// LoraLinear4bit_20675972563492
// MI455X (gfx1250) — matrix operand order repaired and checked by reading, not compiled, not run on hardware
//
#include <hip/hip_runtime.h>

// Problem constants (match reference setup_inputs)
#define TOKENS 256
#define INF    4096
#define OUTF   11008
#define RLORA  16
#define QBLK   64

// Tiling
#define BM 128
#define BN 128
#define BK 32
#define PAD 8
#define THREADS 256

#if defined(__AMDGCN__) && __has_builtin(__builtin_amdgcn_global_load_async_to_lds_b128) && __has_builtin(__builtin_amdgcn_s_wait_asynccnt)
#define USE_ASYNC 1
#else
#define USE_ASYNC 0
#endif

typedef __bf16 bf16x16 __attribute__((ext_vector_type(16)));
typedef __bf16 bf16x2  __attribute__((ext_vector_type(2)));
typedef float  f32x8   __attribute__((ext_vector_type(8)));
typedef int    v4i     __attribute__((ext_vector_type(4)));

#if USE_ASYNC
// global source: AS(1) int4-vector pointer; LDS dest: AS(3) int4-vector pointer
#define GPTR(p) ((__attribute__((address_space(1))) v4i*)(p))
#define LPTR(p) ((__attribute__((address_space(3))) v4i*)(p))
#endif

union FragAB {
  uint4   u[2];
  bf16x16 v;
};
union PK2 {
  unsigned int u;
  bf16x2       v;
};

__device__ __constant__ float NF4C[16] = {
    -1.0f, -0.6961928009986877f, -0.5250730514526367f, -0.39491748809814453f,
    -0.28444138169288635f, -0.18477343022823334f, -0.09105003625154495f, 0.0f,
    0.07958029955625534f, 0.16093020141124725f, 0.24611230194568634f,
    0.33791524171829224f, 0.44070982933044434f, 0.5626170039176941f,
    0.7229568362236023f, 1.0f};

__device__ __forceinline__ unsigned short f2bf(float f) {
  union { float f; unsigned int u; } c;
  c.f = f;
  unsigned int u = c.u;
  // round-to-nearest-even fp32 -> bf16
  return (unsigned short)((u + 0x7fffu + ((u >> 16) & 1u)) >> 16);
}

// ---------------------------------------------------------------------------
// Kernel 1: convert x (f32) -> bf16 bit patterns in workspace
// ---------------------------------------------------------------------------
__global__ __launch_bounds__(256) void cvt_x_kernel(
    const float* __restrict__ x, unsigned short* __restrict__ xb) {
  int i = (blockIdx.x * 256 + threadIdx.x) * 4;
  float4 v = *(const float4*)(x + i);
  uint2 p;
  p.x = (unsigned int)f2bf(v.x) | ((unsigned int)f2bf(v.y) << 16);
  p.y = (unsigned int)f2bf(v.z) | ((unsigned int)f2bf(v.w) << 16);
  *(uint2*)(xb + i) = p;
}

// ---------------------------------------------------------------------------
// Kernel 2: xa[t,r] = 2.0 * sum_k x[t,k] * lora_A[r,k], written as bf16 with
// K padded to 32 (upper 16 lanes zero) so the LoRA GEMM can run on WMMA.
// ---------------------------------------------------------------------------
__global__ __launch_bounds__(256) void xa_kernel(
    const float* __restrict__ x, const float* __restrict__ A,
    unsigned short* __restrict__ xab) {
  const int t = blockIdx.x;
  float acc[RLORA] = {};
  for (int k = threadIdx.x; k < INF; k += 256) {
    float xv = x[(size_t)t * INF + k];
#pragma unroll
    for (int r = 0; r < RLORA; ++r) acc[r] += xv * A[(size_t)r * INF + k];
  }
#pragma unroll
  for (int r = 0; r < RLORA; ++r)
    for (int off = 16; off > 0; off >>= 1)
      acc[r] += __shfl_down(acc[r], off, 32);

  __shared__ float red[8][RLORA];
  const int lane = threadIdx.x & 31, wv = threadIdx.x >> 5;
  if (lane == 0) {
#pragma unroll
    for (int r = 0; r < RLORA; ++r) red[wv][r] = acc[r];
  }
  __syncthreads();
  if (threadIdx.x < 2 * RLORA) {
    unsigned short v = 0;
    if (threadIdx.x < RLORA) {
      float s = 0.f;
#pragma unroll
      for (int w = 0; w < 8; ++w) s += red[w][threadIdx.x];
      v = f2bf(2.0f * s);  // lora_alpha / r folded in
    }
    xab[(size_t)t * (2 * RLORA) + threadIdx.x] = v;
  }
}

// ---------------------------------------------------------------------------
// Kernel 3: convert lora_B [OUTF,16] f32 -> bf16 rows padded to K=32
// ---------------------------------------------------------------------------
__global__ __launch_bounds__(256) void cvt_b_kernel(
    const float* __restrict__ B, unsigned short* __restrict__ Bb) {
  int n = blockIdx.x * 256 + threadIdx.x;  // OUTF = 43 * 256 exactly
  const float4* p = (const float4*)(B + (size_t)n * RLORA);
  float4 b0 = p[0], b1 = p[1], b2 = p[2], b3 = p[3];
  unsigned int w[8];
  w[0] = (unsigned int)f2bf(b0.x) | ((unsigned int)f2bf(b0.y) << 16);
  w[1] = (unsigned int)f2bf(b0.z) | ((unsigned int)f2bf(b0.w) << 16);
  w[2] = (unsigned int)f2bf(b1.x) | ((unsigned int)f2bf(b1.y) << 16);
  w[3] = (unsigned int)f2bf(b1.z) | ((unsigned int)f2bf(b1.w) << 16);
  w[4] = (unsigned int)f2bf(b2.x) | ((unsigned int)f2bf(b2.y) << 16);
  w[5] = (unsigned int)f2bf(b2.z) | ((unsigned int)f2bf(b2.w) << 16);
  w[6] = (unsigned int)f2bf(b3.x) | ((unsigned int)f2bf(b3.y) << 16);
  w[7] = (unsigned int)f2bf(b3.z) | ((unsigned int)f2bf(b3.w) << 16);
  unsigned short* dst = Bb + (size_t)n * 32;
  *(uint4*)(dst + 0)  = make_uint4(w[0], w[1], w[2], w[3]);
  *(uint4*)(dst + 8)  = make_uint4(w[4], w[5], w[6], w[7]);
  *(uint4*)(dst + 16) = make_uint4(0u, 0u, 0u, 0u);
  *(uint4*)(dst + 24) = make_uint4(0u, 0u, 0u, 0u);
}

// ---------------------------------------------------------------------------
// Shared fragment-load + 8x WMMA step (per wave), contiguous-k A/B layouts (k = 16*lhalf + s).
// ---------------------------------------------------------------------------
__device__ __forceinline__ void mma_step(
    const unsigned short (*__restrict__ xt)[BK + PAD],
    const unsigned short (*__restrict__ wt)[BK + PAD],
    int wm, int wn, int lhalf, int l15, f32x8 acc[4][2]) {
  FragAB bfr[2];
#pragma unroll
  for (int nf = 0; nf < 2; ++nf) {
    // B (K x N): lane<16 -> K 0..15, lane>=16 -> K 16..31, contiguous 32B
    const uint4* p = (const uint4*)&wt[wn * 32 + nf * 16 + l15][lhalf * 16];
    bfr[nf].u[0] = p[0];
    bfr[nf].u[1] = p[1];
  }
  FragAB afr[4];
#pragma unroll
  for (int mf = 0; mf < 4; ++mf) {
    // A (M x K): lane<16 -> K 0..15, lane>=16 -> K 16..31, contiguous 32B
    const unsigned short* r = &xt[wm * 64 + mf * 16 + l15][0];
    afr[mf].u[0] = *(const uint4*)(r + lhalf * 16);
    afr[mf].u[1] = *(const uint4*)(r + lhalf * 16 + 8);
  }
#pragma unroll
  for (int mf = 0; mf < 4; ++mf)
#pragma unroll
    for (int nf = 0; nf < 2; ++nf)
      acc[mf][nf] = __builtin_amdgcn_wmma_f32_16x16x32_bf16(
          false, afr[mf].v, false, bfr[nf].v, (short)0, acc[mf][nf],
          false, false);
}

// ---------------------------------------------------------------------------
// Kernel 4: fused NF4-dequant GEMM, LoRA folded in as one extra WMMA K-step.
// Block = 128(M) x 128(N), 8 waves 2x4, each wave 64x32, K-step 32,
// double-buffered LDS (one barrier per K-step).
// ---------------------------------------------------------------------------
__global__ __launch_bounds__(THREADS) void nf4_lora_gemm(
    const unsigned short* __restrict__ xb, const int* __restrict__ qc,
    const float* __restrict__ absmax, const unsigned short* __restrict__ bb,
    const unsigned short* __restrict__ xab, float* __restrict__ out) {
  __shared__ __align__(16) unsigned short xtile[2][BM][BK + PAD];
  __shared__ __align__(16) unsigned short wtile[2][BN][BK + PAD];
  __shared__ unsigned short nf4tb[16];

  const int tix = threadIdx.x;
  const int n0 = blockIdx.x * BN;
  const int m0 = blockIdx.y * BM;

  if (tix < 16) nf4tb[tix] = f2bf(NF4C[tix]);

  const int lane  = tix & 31;
  const int wv    = tix >> 5;
  const int wm    = wv >> 2;   // 0..1
  const int wn    = wv & 3;    // 0..3
  const int lhalf = lane >> 4;
  const int l15   = lane & 15;

  // cooperative-load roles: each thread owns 16 contiguous k of one row
  const int rw = tix >> 1;  // 0..127
  const int kc = tix & 1;   // 0..1
  const unsigned short* xrow = xb + (size_t)(m0 + rw) * INF + kc * 16;
  const int* qrow = qc + (size_t)(n0 + rw) * INF + kc * 16;
  const float* amrow = absmax + (size_t)(n0 + rw) * (INF / QBLK);

  f32x8 acc[4][2] = {};
  __syncthreads();  // nf4tb visible

  int buf = 0;
  for (int ks = 0; ks < INF; ks += BK, buf ^= 1) {
    unsigned short(*xt)[BK + PAD] = xtile[buf];
    unsigned short(*wt)[BK + PAD] = wtile[buf];

    // ---- x tile: global -> LDS ---------------------------------------
#if USE_ASYNC
    __builtin_amdgcn_global_load_async_to_lds_b128(
        GPTR(xrow + ks), LPTR(&xt[rw][kc * 16]), 0, 0);
    __builtin_amdgcn_global_load_async_to_lds_b128(
        GPTR(xrow + ks + 8), LPTR(&xt[rw][kc * 16 + 8]), 0, 0);
#else
    uint4 xv0 = *(const uint4*)(xrow + ks);
    uint4 xv1 = *(const uint4*)(xrow + ks + 8);
    *(uint4*)&xt[rw][kc * 16]     = xv0;
    *(uint4*)&xt[rw][kc * 16 + 8] = xv1;
#endif

    // ---- q codes + dequant (packed bf16 multiply) --------------------
    int4 q0 = *(const int4*)(qrow + ks);
    int4 q1 = *(const int4*)(qrow + ks + 4);
    int4 q2 = *(const int4*)(qrow + ks + 8);
    int4 q3 = *(const int4*)(qrow + ks + 12);
    float am = amrow[(ks + kc * 16) >> 6];
    if (ks + BK < INF) {
      __builtin_prefetch(qrow + ks + BK, 0, 1);  // global_prefetch_b8
      __builtin_prefetch(xrow + ks + BK, 0, 1);
    }

    int codes[16] = {q0.x, q0.y, q0.z, q0.w, q1.x, q1.y, q1.z, q1.w,
                     q2.x, q2.y, q2.z, q2.w, q3.x, q3.y, q3.z, q3.w};
    unsigned short amh = f2bf(am);
    PK2 amv;
    amv.u = (unsigned int)amh | ((unsigned int)amh << 16);
    unsigned int pk[8];
#pragma unroll
    for (int j = 0; j < 8; ++j) {
      PK2 t;
      t.u = (unsigned int)nf4tb[codes[2 * j]] |
            ((unsigned int)nf4tb[codes[2 * j + 1]] << 16);
      t.v = t.v * amv.v;  // -> v_pk_mul_bf16
      pk[j] = t.u;
    }
    *(uint4*)&wt[rw][kc * 16]     = make_uint4(pk[0], pk[1], pk[2], pk[3]);
    *(uint4*)&wt[rw][kc * 16 + 8] = make_uint4(pk[4], pk[5], pk[6], pk[7]);

#if USE_ASYNC
    __builtin_amdgcn_s_wait_asynccnt(0);
#endif
    __syncthreads();

    mma_step(xt, wt, wm, wn, lhalf, l15, acc);
  }

  // ---- LoRA as one extra WMMA K-step (K=16 padded to 32) --------------
  {
    unsigned short(*xt)[BK + PAD] = xtile[0];
    unsigned short(*wt)[BK + PAD] = wtile[0];
    const unsigned short* xar = xab + (size_t)(m0 + rw) * 32 + kc * 16;
    const unsigned short* bbr = bb + (size_t)(n0 + rw) * 32 + kc * 16;
    *(uint4*)&xt[rw][kc * 16]     = *(const uint4*)xar;
    *(uint4*)&xt[rw][kc * 16 + 8] = *(const uint4*)(xar + 8);
    *(uint4*)&wt[rw][kc * 16]     = *(const uint4*)bbr;
    *(uint4*)&wt[rw][kc * 16 + 8] = *(const uint4*)(bbr + 8);
    __syncthreads();
    mma_step(xt, wt, wm, wn, lhalf, l15, acc);
  }

  // ---- store D tiles (C layout: N = l15, M = lhalf*8 + e) -------------
#pragma unroll
  for (int nf = 0; nf < 2; ++nf) {
    const int ng = n0 + wn * 32 + nf * 16 + l15;
#pragma unroll
    for (int mf = 0; mf < 4; ++mf) {
#pragma unroll
      for (int e = 0; e < 8; ++e) {
        const int tl = wm * 64 + mf * 16 + lhalf * 8 + e;
        out[(size_t)(m0 + tl) * OUTF + ng] = acc[mf][nf][e];
      }
    }
  }
}

// ---------------------------------------------------------------------------
extern "C" void kernel_launch(void* const* d_in, const int* in_sizes, int n_in,
                              void* d_out, int out_size, void* d_ws,
                              size_t ws_size, hipStream_t stream) {
  (void)in_sizes; (void)n_in; (void)out_size; (void)ws_size;
  const float* x      = (const float*)d_in[0];
  const int*   qc     = (const int*)d_in[1];
  const float* absmax = (const float*)d_in[2];
  const float* lora_A = (const float*)d_in[3];
  const float* lora_B = (const float*)d_in[4];
  float* out = (float*)d_out;

  // workspace layout: xb (2 MB) | xab (16 KB) | Bb (704 KB)
  unsigned short* xb  = (unsigned short*)d_ws;
  unsigned short* xab = (unsigned short*)((char*)d_ws + (size_t)TOKENS * INF * 2);
  unsigned short* Bb  = (unsigned short*)((char*)d_ws + (size_t)TOKENS * INF * 2 +
                                          (size_t)TOKENS * 32 * 2);

  cvt_x_kernel<<<(TOKENS * INF) / (256 * 4), 256, 0, stream>>>(x, xb);
  xa_kernel<<<TOKENS, 256, 0, stream>>>(x, lora_A, xab);
  cvt_b_kernel<<<OUTF / 256, 256, 0, stream>>>(lora_B, Bb);
  nf4_lora_gemm<<<dim3(OUTF / BN, TOKENS / BM), THREADS, 0, stream>>>(
      xb, qc, absmax, Bb, xab, out);
}
